// MultiHeadedAttention_6339371729780
// MI455X (gfx1250) — compile-verified, matrix operand order checked by reading, not run on hardware
//
#include <hip/hip_runtime.h>

typedef __attribute__((ext_vector_type(16))) __bf16 v16bf;
typedef __attribute__((ext_vector_type(8)))  float  v8f;
typedef __attribute__((ext_vector_type(4)))  float  f4v;
typedef __attribute__((ext_vector_type(8)))  unsigned short ushort8;

#define S_LEN   4096
#define DMODEL  512
#define NHEAD   8
#define DKDIM   64
#define M_TOT   8192            /* B * S */
#define MWORDS  (S_LEN / 32)    /* 128 mask words per row */

union FragBF {
  v16bf v;
  ushort8 h[2];
  unsigned short u[16];
};

__device__ __forceinline__ unsigned short f2bf(float f) {
  union { float f; unsigned u; } x; x.f = f;
  unsigned r = x.u + 0x7fffu + ((x.u >> 16) & 1u);   // round-to-nearest-even
  return (unsigned short)(r >> 16);
}

// bf16 A/B fragment: two contiguous 16B chunks per lane
__device__ __forceinline__ v16bf load_frag(const unsigned short* row, int k0, int koff) {
  FragBF f;
  f.h[0] = *(const ushort8*)(row + k0 + koff);
  f.h[1] = *(const ushort8*)(row + k0 + koff + 16);
  return f.v;
}

__device__ __forceinline__ v8f wmma_bf16(v16bf a, v16bf b, v8f c) {
  return __builtin_amdgcn_wmma_f32_16x16x32_bf16(false, a, false, b, (short)0, c,
                                                 false, false);
}

// ---------------------------------------------------------------------------
// Prepack: fp32 -> bf16, 8 elements per thread (pure b128 in / b128 out)
// ---------------------------------------------------------------------------
__global__ __launch_bounds__(256) void cvt_bf16_kernel(const float* __restrict__ src,
                                                       unsigned short* __restrict__ dst) {
  const size_t base = ((size_t)blockIdx.x * 256 + threadIdx.x) * 8;
  f4v x0 = *(const f4v*)(src + base);
  f4v x1 = *(const f4v*)(src + base + 4);
  ushort8 o;
#pragma unroll
  for (int i = 0; i < 4; ++i) { o[i] = f2bf(x0[i]); o[i + 4] = f2bf(x1[i]); }
  *(ushort8*)(dst + base) = o;
}

// ---------------------------------------------------------------------------
// Prepack: int mask -> bit mask via wave32 ballot (134 MB read once; 4 MB out)
// ---------------------------------------------------------------------------
__global__ __launch_bounds__(256) void pack_mask_kernel(const int* __restrict__ mask,
                                                        unsigned* __restrict__ mbits) {
  const size_t gid = (size_t)blockIdx.x * 256 + threadIdx.x;
  const unsigned long long bal = __ballot(mask[gid] != 0);
  if ((threadIdx.x & 31) == 0) mbits[gid >> 5] = (unsigned)bal;
}

// ---------------------------------------------------------------------------
// GEMM: C[m,n] = scale * sum_k A[m,k] * W[n,k]   (A,W bf16; M=8192,N=K=512)
// MODE 0: bf16 row-major [M,512]   MODE 1: bf16 Vt[(b*512+n)*4096+s]
// MODE 2: fp32 row-major [M,512]
// 256 threads = 8 waves; block tile 128x128; wave tile 32x64 (8 WMMA/k-step).
// ---------------------------------------------------------------------------
template <int MODE>
__global__ __launch_bounds__(256) void gemm_wmma_kernel(
    const unsigned short* __restrict__ A, const unsigned short* __restrict__ W,
    void* __restrict__ outp, float scale) {
  const int lane = threadIdx.x & 31;
  const int wave = threadIdx.x >> 5;
  const int col  = lane & 15;
  const int koff = (lane & 16) >> 1;          // 0 or 8 (also C-row offset)
  const int bm = (blockIdx.x & 63) * 128;     // M/128 = 64
  const int bn = (blockIdx.x >> 6) * 128;     // N/128 = 4
  const int wm = bm + (wave >> 1) * 32;
  const int wn = bn + (wave & 1) * 64;

  v8f c[2][4] = {};

  const unsigned short* a0p = A + (size_t)(wm + col) * DMODEL;
  const unsigned short* a1p = a0p + 16 * DMODEL;
  const unsigned short* wp[4];
#pragma unroll
  for (int tj = 0; tj < 4; ++tj) wp[tj] = W + (size_t)(wn + tj * 16 + col) * DMODEL;

  for (int k0 = 0; k0 < DMODEL; k0 += 32) {
    v16bf a0 = load_frag(a0p, k0, koff);
    v16bf a1 = load_frag(a1p, k0, koff);
#pragma unroll
    for (int tj = 0; tj < 4; ++tj) {
      v16bf bf = load_frag(wp[tj], k0, koff);
      c[0][tj] = wmma_bf16(a0, bf, c[0][tj]);
      c[1][tj] = wmma_bf16(a1, bf, c[1][tj]);
    }
  }

#pragma unroll
  for (int ti = 0; ti < 2; ++ti) {
#pragma unroll
    for (int tj = 0; tj < 4; ++tj) {
      const int mbase = wm + ti * 16 + koff;
      const int n     = wn + tj * 16 + col;
      if (MODE == 0) {
        unsigned short* Cb = (unsigned short*)outp;
#pragma unroll
        for (int r = 0; r < 8; ++r)
          Cb[(size_t)(mbase + r) * DMODEL + n] = f2bf(c[ti][tj][r] * scale);
      } else if (MODE == 1) {
        unsigned short* Vt = (unsigned short*)outp;
        const int b = mbase >> 12;
        const int s = mbase & (S_LEN - 1);
        ushort8 pk;
#pragma unroll
        for (int r = 0; r < 8; ++r) pk[r] = f2bf(c[ti][tj][r] * scale);
        *(ushort8*)(Vt + ((size_t)b * DMODEL + n) * S_LEN + s) = pk;
      } else {
        float* Cf = (float*)outp;
#pragma unroll
        for (int r = 0; r < 8; ++r)
          Cf[(size_t)(mbase + r) * DMODEL + n] = c[ti][tj][r] * scale;
      }
    }
  }
}

// ---------------------------------------------------------------------------
// Flash attention: each wave owns 16 query rows of one (b,h).
// Qb (prescaled by 1/8), Kb: bf16 [B,S,512]; Vt: bf16 [B,H,64,S].
// mbits[b][q][k/32]: bit set -> masked -> score = -1e9.
// ---------------------------------------------------------------------------
__global__ __launch_bounds__(128) void flash_attn_kernel(
    const unsigned short* __restrict__ Qb, const unsigned short* __restrict__ Kb,
    const unsigned short* __restrict__ Vt, const unsigned* __restrict__ mbits,
    unsigned short* __restrict__ Ob) {
  __shared__ unsigned short plds[4][16][32];   // per-wave P relayout buffer
  const int lane = threadIdx.x & 31;
  const int wave = threadIdx.x >> 5;
  const int col  = lane & 15;
  const int hi   = (lane & 16) >> 1;           // 0 or 8
  const int tile = blockIdx.x * 4 + wave;      // 4096 tiles
  const int qt = tile & (S_LEN / 16 - 1);
  const int bh = tile >> 8;
  const int b = bh >> 3, h = bh & 7;
  const int s0 = qt * 16;

  const unsigned short* qrow = Qb + ((size_t)b * S_LEN + s0 + col) * DMODEL + h * DKDIM;
  const v16bf qa0 = load_frag(qrow, 0, hi);
  const v16bf qa1 = load_frag(qrow, 32, hi);

  const unsigned short* kbase = Kb + (size_t)b * S_LEN * DMODEL + h * DKDIM;
  const unsigned short* vbase = Vt + ((size_t)b * DMODEL + h * DKDIM) * S_LEN;
  const unsigned* mwrow = mbits + ((size_t)b * S_LEN + s0 + col) * MWORDS;  // lane's row

  v8f acc[4] = {};
  float mr[8], lr[8];
#pragma unroll
  for (int r = 0; r < 8; ++r) { mr[r] = -3.0e38f; lr[r] = 0.0f; }

  for (int kb = 0; kb < S_LEN; kb += 32) {
    // ---- scores (Q prescaled): two 16x16 tiles over this 32-key chunk
    const unsigned short* k0row = kbase + (size_t)(kb + col) * DMODEL;
    const unsigned short* k1row = k0row + 16 * DMODEL;
    v8f st0 = {}, st1 = {};
    st0 = wmma_bf16(qa0, load_frag(k0row, 0, hi),  st0);
    st0 = wmma_bf16(qa1, load_frag(k0row, 32, hi), st0);
    st1 = wmma_bf16(qa0, load_frag(k1row, 0, hi),  st1);
    st1 = wmma_bf16(qa1, load_frag(k1row, 32, hi), st1);

    // ---- mask from packed bits: 1 load/lane, distribute rows via shfl
    const unsigned wbits = mwrow[kb >> 5];
    float p0[8], p1[8];
#pragma unroll
    for (int r = 0; r < 8; ++r) {
      const unsigned wr = (unsigned)__shfl((int)wbits, r + hi, 32);
      float x0 = st0[r];
      float x1 = st1[r];
      if ((wr >> col) & 1u)        x0 = -1.0e9f;
      if ((wr >> (col + 16)) & 1u) x1 = -1.0e9f;
      p0[r] = x0; p1[r] = x1;
    }

    // ---- online softmax (row-wise across 16 lanes of each half)
#pragma unroll
    for (int r = 0; r < 8; ++r) {
      float mx = fmaxf(p0[r], p1[r]);
      mx = fmaxf(mx, __shfl_xor(mx, 1, 32));
      mx = fmaxf(mx, __shfl_xor(mx, 2, 32));
      mx = fmaxf(mx, __shfl_xor(mx, 4, 32));
      mx = fmaxf(mx, __shfl_xor(mx, 8, 32));
      const float mnew  = fmaxf(mr[r], mx);
      const float alpha = __expf(mr[r] - mnew);
      mr[r] = mnew;
      p0[r] = __expf(p0[r] - mnew);
      p1[r] = __expf(p1[r] - mnew);
      lr[r] = lr[r] * alpha + p0[r] + p1[r];
#pragma unroll
      for (int t = 0; t < 4; ++t) acc[t][r] *= alpha;
      // C-layout -> LDS row-major P[16][32] (wave-private; DS in-order per wave)
      plds[wave][r + hi][col]      = f2bf(p0[r]);
      plds[wave][r + hi][col + 16] = f2bf(p1[r]);
    }

    // ---- reload P as an A-fragment (16x32 bf16)
    FragBF pf;
    const unsigned short* prow = &plds[wave][col][0];
    pf.h[0] = *(const ushort8*)(prow + hi);
    pf.h[1] = *(const ushort8*)(prow + hi + 16);

    // ---- P (16x32) x V (32x64): 4 WMMA against Vt fragments
#pragma unroll
    for (int t = 0; t < 4; ++t) {
      const unsigned short* vrow = vbase + (size_t)(t * 16 + col) * S_LEN + kb;
      acc[t] = wmma_bf16(pf.v, load_frag(vrow, 0, hi), acc[t]);
    }
  }

  // ---- finalize: reduce l, normalize, store bf16 [B,S,512]
  float inv[8];
#pragma unroll
  for (int r = 0; r < 8; ++r) {
    float l = lr[r];
    l += __shfl_xor(l, 1, 32);
    l += __shfl_xor(l, 2, 32);
    l += __shfl_xor(l, 4, 32);
    l += __shfl_xor(l, 8, 32);
    inv[r] = 1.0f / l;
  }
#pragma unroll
  for (int t = 0; t < 4; ++t) {
#pragma unroll
    for (int r = 0; r < 8; ++r) {
      Ob[((size_t)b * S_LEN + s0 + hi + r) * DMODEL + h * DKDIM + t * 16 + col] =
          f2bf(acc[t][r] * inv[r]);
    }
  }
}

// ---------------------------------------------------------------------------
extern "C" void kernel_launch(void* const* d_in, const int* in_sizes, int n_in,
                              void* d_out, int out_size, void* d_ws, size_t ws_size,
                              hipStream_t stream) {
  (void)in_sizes; (void)n_in; (void)out_size; (void)ws_size;
  const float* query = (const float*)d_in[0];
  const float* key   = (const float*)d_in[1];
  const float* value = (const float*)d_in[2];
  const int*   mask  = (const int*)d_in[3];
  const float* Wq    = (const float*)d_in[4];
  const float* Wk    = (const float*)d_in[5];
  const float* Wv    = (const float*)d_in[6];
  const float* Wo    = (const float*)d_in[7];
  float* out = (float*)d_out;

  char* ws = (char*)d_ws;
  const size_t ACT = (size_t)M_TOT * DMODEL * sizeof(unsigned short);   // 8 MiB
  const size_t WSZ = (size_t)DMODEL * DMODEL * sizeof(unsigned short);  // 512 KiB
  unsigned short* Qb  = (unsigned short*)(ws);
  unsigned short* Kb  = (unsigned short*)(ws + ACT);
  unsigned short* Vt  = (unsigned short*)(ws + 2 * ACT);
  unsigned short* Ab  = (unsigned short*)(ws + 3 * ACT);
  unsigned short* Qin = (unsigned short*)(ws + 4 * ACT);
  unsigned short* Kin = (unsigned short*)(ws + 5 * ACT);
  unsigned short* Vin = (unsigned short*)(ws + 6 * ACT);
  unsigned short* Wqb = (unsigned short*)(ws + 7 * ACT);
  unsigned short* Wkb = (unsigned short*)(ws + 7 * ACT + WSZ);
  unsigned short* Wvb = (unsigned short*)(ws + 7 * ACT + 2 * WSZ);
  unsigned short* Wob = (unsigned short*)(ws + 7 * ACT + 3 * WSZ);
  unsigned*       Mb  = (unsigned*)(ws + 7 * ACT + 4 * WSZ);            // 4 MiB

  // ---- prepack: mask bits + bf16 activations/weights
  pack_mask_kernel<<<131072, 256, 0, stream>>>(mask, Mb);  // 2*4096*4096 / 256
  cvt_bf16_kernel<<<2048, 256, 0, stream>>>(query, Qin);   // 4.19M elems / 8 / 256
  cvt_bf16_kernel<<<2048, 256, 0, stream>>>(key,   Kin);
  cvt_bf16_kernel<<<2048, 256, 0, stream>>>(value, Vin);
  cvt_bf16_kernel<<<128, 256, 0, stream>>>(Wq, Wqb);       // 262144 / 8 / 256
  cvt_bf16_kernel<<<128, 256, 0, stream>>>(Wk, Wkb);
  cvt_bf16_kernel<<<128, 256, 0, stream>>>(Wv, Wvb);
  cvt_bf16_kernel<<<128, 256, 0, stream>>>(Wo, Wob);

  // ---- projections (Q carries the 1/sqrt(d_k) scale; V stored transposed)
  gemm_wmma_kernel<0><<<256, 256, 0, stream>>>(Qin, Wqb, (void*)Qb, 0.125f);
  gemm_wmma_kernel<0><<<256, 256, 0, stream>>>(Kin, Wkb, (void*)Kb, 1.0f);
  gemm_wmma_kernel<1><<<256, 256, 0, stream>>>(Vin, Wvb, (void*)Vt, 1.0f);

  // ---- attention
  flash_attn_kernel<<<1024, 128, 0, stream>>>(Qb, Kb, Vt, Mb, Ab);

  // ---- output projection
  gemm_wmma_kernel<2><<<256, 256, 0, stream>>>(Ab, Wob, (void*)out, 1.0f);
}
